// ReckoningLayer_23948737642948
// MI455X (gfx1250) — matrix operand order repaired and checked by reading, not compiled, not run on hardware
//
#include <hip/hip_runtime.h>
#include <hip/hip_bf16.h>

// ---------------------------------------------------------------------------
// Problem constants
// ---------------------------------------------------------------------------
#define B_ 4
#define T_ 2048
#define D_ 1024
#define DM_ 256
#define DF_ 4096
#define KC_ 7

typedef unsigned short u16;
typedef unsigned int u32;

typedef __attribute__((ext_vector_type(16))) __bf16 v16bf;
typedef __attribute__((ext_vector_type(8)))  float  v8f;

union Frag { u32 u[8]; v16bf v; };

static __device__ __forceinline__ u16 f2bf(float f) {
    union { float f; u32 u; } x; x.f = f;
    u32 r = x.u + 0x7FFFu + ((x.u >> 16) & 1u);   // RNE
    return (u16)(r >> 16);
}
static __device__ __forceinline__ float sigm(float x) {
    return 1.0f / (1.0f + __expf(-x));
}
static __device__ __forceinline__ v8f vzero() {
    v8f z = {0.f,0.f,0.f,0.f,0.f,0.f,0.f,0.f}; return z;
}

// ---------------------------------------------------------------------------
// CDNA5 async global->LDS copy (ASYNCcnt-tracked, writes LDS directly).
// lds_off: byte offset within the workgroup LDS allocation (low 32 bits of a
// generic __shared__ pointer). Each lane copies 16 contiguous bytes.
// ---------------------------------------------------------------------------
static __device__ __forceinline__ void async_g2l_b128(u32 lds_off, const u16* g) {
    asm volatile("global_load_async_to_lds_b128 %0, %1, off"
                 :: "v"(lds_off), "v"(g) : "memory");
}
#define WAIT_ASYNCCNT(n) asm volatile("s_wait_asynccnt %0" :: "i"(n) : "memory")

// ---------------------------------------------------------------------------
// Generic f32 -> bf16 convert
// ---------------------------------------------------------------------------
__global__ void cvt_bf16_kernel(const float* __restrict__ in, u16* __restrict__ out, int n) {
    int i = blockIdx.x * blockDim.x + threadIdx.x;
    if (i < n) out[i] = f2bf(in[i]);
}

// ---------------------------------------------------------------------------
// RMSNorm over D=1024, one block (256 thr) per token. Optional f32/bf16 out.
// ---------------------------------------------------------------------------
__global__ __launch_bounds__(256) void rmsnorm_kernel(
    const float* __restrict__ x, const float* __restrict__ w,
    float* __restrict__ outF, u16* __restrict__ outB)
{
    __shared__ float red[256];
    const size_t tok = blockIdx.x;
    const float* xr = x + tok * D_;
    int tid = threadIdx.x;
    float ss = 0.f;
    #pragma unroll
    for (int i = 0; i < 4; ++i) { float v = xr[tid + i*256]; ss += v*v; }
    red[tid] = ss; __syncthreads();
    for (int s = 128; s > 0; s >>= 1) {
        if (tid < s) red[tid] += red[tid + s];
        __syncthreads();
    }
    float rms = rsqrtf(red[0] * (1.0f / D_) + 1e-6f);
    #pragma unroll
    for (int i = 0; i < 4; ++i) {
        int c = tid + i*256;
        float v = xr[c] * rms * w[c];
        if (outF) outF[tok*D_ + c] = v;
        if (outB) outB[tok*D_ + c] = f2bf(v);
    }
}

// ---------------------------------------------------------------------------
// Causal depthwise conv, K=7, left pad 6 (cross-correlation).
// ---------------------------------------------------------------------------
__global__ void dwconv_kernel(const float* __restrict__ h, const float* __restrict__ w,
                              float* __restrict__ out)
{
    size_t idx = (size_t)blockIdx.x * blockDim.x + threadIdx.x;
    if (idx >= (size_t)B_*T_*D_) return;
    int d = idx % D_;
    int t = (idx / D_) % T_;
    size_t b = idx / ((size_t)D_*T_);
    float acc = 0.f;
    #pragma unroll
    for (int j = 0; j < KC_; ++j) {
        int tt = t - (KC_-1) + j;
        if (tt >= 0) acc += h[(b*T_ + tt)*D_ + d] * w[d*KC_ + j];
    }
    out[idx] = acc;
}

// ---------------------------------------------------------------------------
// Gated linear state recurrence: one thread per (b,d) chain, loop over T.
// ---------------------------------------------------------------------------
__global__ void state_scan_kernel(const float* __restrict__ dpre, const float* __restrict__ gpre,
                                  const float* __restrict__ h, const float* __restrict__ sscale,
                                  float* __restrict__ state)
{
    int tid = blockIdx.x * blockDim.x + threadIdx.x;   // 0..4095
    int b = tid >> 10, d = tid & 1023;
    float s = 0.f, sc = *sscale;
    for (int t = 0; t < T_; ++t) {
        size_t idx = ((size_t)(b*T_ + t))*D_ + d;
        float dk = sigm(dpre[idx]);
        dk = fminf(fmaxf(dk, 0.01f), 0.99f);
        float u = sigm(gpre[idx]) * h[idx];
        s = dk * s + u;
        state[idx] = s * sc;
    }
}

// ---------------------------------------------------------------------------
// beta[b,t] = sigmoid(h[b,t,:] . Wbeta) ; one wave32 per token.
// ---------------------------------------------------------------------------
__global__ __launch_bounds__(256) void beta_kernel(const float* __restrict__ h,
                                                   const float* __restrict__ wb,
                                                   float* __restrict__ beta)
{
    int wid = (blockIdx.x * blockDim.x + threadIdx.x) >> 5;
    int lane = threadIdx.x & 31;
    if (wid >= B_*T_) return;
    const float* hr = h + (size_t)wid * D_;
    float s = 0.f;
    for (int j = lane; j < D_; j += 32) s += hr[j] * wb[j];
    #pragma unroll
    for (int off = 16; off > 0; off >>= 1) s += __shfl_xor(s, off, 32);
    if (lane == 0) beta[wid] = sigm(s);
}

// ---------------------------------------------------------------------------
// DeltaNet rank-1 memory scan. M (256x256 f32, pitch 257) lives in LDS
// (263KB < 320KB WGP LDS). One workgroup of 1024 threads = (b, i) pairs.
// Writes mem_out directly as bf16 for the following Wmem GEMM.
// ---------------------------------------------------------------------------
#define MEMSCAN_LDS_FLOATS (256*257 + 1024 + 1024 + 256 + 256 + 8)
__global__ __launch_bounds__(1024) void memscan_kernel(
    const float* __restrict__ M0, const float* __restrict__ kf,
    const float* __restrict__ vf, const float* __restrict__ betaf,
    u16* __restrict__ memb)
{
    extern __shared__ float sm[];
    float* Ml   = sm;                 // 256*257
    float* kv   = Ml + 256*257;       // 4*256
    float* bcs  = kv + 1024;          // 4*256
    float* ubuf = bcs + 1024;         // 256
    float* kbar = ubuf + 256;         // 256
    float* bet  = kbar + 256;         // 4 (+pad)

    int tid = threadIdx.x;
    int b = tid >> 8, i = tid & 255;
    // Load M0 (row pitch 257 -> conflict-free column access across lanes)
    for (int e = tid; e < 256*256; e += 1024)
        Ml[(e >> 8)*257 + (e & 255)] = M0[e];
    __syncthreads();

    for (int t = 0; t < T_; ++t) {
        size_t idx = ((size_t)(b*T_ + t))*DM_ + i;
        kv[b*256 + i] = kf[idx];
        float vreg = vf[idx];
        if (tid < 4) bet[tid] = betaf[tid*T_ + t];
        __syncthreads();

        // read[b][i] = M[i,:] . k[b]
        const float* Mr = &Ml[i*257];
        const float* kb = &kv[b*256];
        float r = 0.f;
        #pragma unroll 8
        for (int j = 0; j < 256; ++j) r += Mr[j] * kb[j];
        memb[idx] = f2bf(r);
        bcs[b*256 + i] = bet[b] * (vreg - r);
        __syncthreads();

        if (tid < 256) {
            ubuf[tid] = 0.25f * (bcs[tid] + bcs[256+tid] + bcs[512+tid] + bcs[768+tid]);
        } else if (tid < 512) {
            int j = tid - 256;
            kbar[j] = 0.25f * (kv[j] + kv[256+j] + kv[512+j] + kv[768+j]);
        }
        __syncthreads();

        // M += outer(u, kbar): 64 elements per thread, coalesced in j
        for (int l = 0; l < 64; ++l) {
            int e = tid + l*1024;
            int ii = e >> 8, jj = e & 255;
            Ml[ii*257 + jj] += ubuf[ii] * kbar[jj];
        }
        __syncthreads();
    }
}

// ---------------------------------------------------------------------------
// Pack [local | state | mem] (f32) -> bf16 concat buffer [B,T,3D]
// ---------------------------------------------------------------------------
__global__ void concat_kernel(const float* __restrict__ loc, const float* __restrict__ st,
                              const float* __restrict__ mem, u16* __restrict__ cat)
{
    size_t idx = (size_t)blockIdx.x * blockDim.x + threadIdx.x;
    if (idx >= (size_t)B_*T_*3*D_) return;
    int c = idx % (3*D_);
    size_t bt = idx / (3*D_);
    float v;
    if (c < D_)            v = loc[bt*D_ + c];
    else if (c < 2*D_)     v = st [bt*D_ + (c - D_)];
    else                   v = mem[bt*D_ + (c - 2*D_)];
    cat[idx] = f2bf(v);
}

// ---------------------------------------------------------------------------
// bf16 WMMA GEMM: C[M,N] = scale * (A[M,K] @ W[N,K]^T) + bias + resid
// Block tile 128x64, BK=64, 8 waves (4x2), wave tile 32x32 (4 wmma accums).
// Double-buffered LDS tiles streamed with global_load_async_to_lds_b128 and
// pipelined with s_wait_asynccnt (in-order async completion).
// ---------------------------------------------------------------------------
#define BM 128
#define BN 64
#define BK 64
#define PITCH16 72   // u16 per LDS row: 64 data + 8 pad (144 B = 9 x 16 B)
#define PITCH32 36

static __device__ __forceinline__ void issue_tile_A(
    const u16* __restrict__ A, int row0, int K, int kt, u32 base, int tid)
{
    #pragma unroll
    for (int l = 0; l < 4; ++l) {           // 128 rows x 8 chunks = 1024 / 256 thr
        int e = tid + l*256;
        int ar = e >> 3, c = e & 7;
        async_g2l_b128(base + ar*(PITCH16*2) + c*16,
                       A + (size_t)(row0 + ar)*K + kt + c*8);
    }
}
static __device__ __forceinline__ void issue_tile_B(
    const u16* __restrict__ W, int col0, int K, int kt, u32 base, int tid)
{
    #pragma unroll
    for (int l = 0; l < 2; ++l) {           // 64 rows x 8 chunks = 512 / 256 thr
        int e = tid + l*256;
        int br = e >> 3, c = e & 7;
        async_g2l_b128(base + br*(PITCH16*2) + c*16,
                       W + (size_t)(col0 + br)*K + kt + c*8);
    }
}

__global__ __launch_bounds__(256) void gemm_bf16_kernel(
    const u16* __restrict__ A, const u16* __restrict__ W,
    float* __restrict__ Cf, u16* __restrict__ Cb,
    const float* __restrict__ bias, const float* __restrict__ resid,
    const float* __restrict__ scale_ptr, int M, int N, int K)
{
    __shared__ u16 AsBuf[2][BM * PITCH16];
    __shared__ u16 BsBuf[2][BN * PITCH16];

    const int tid = threadIdx.x, lane = tid & 31, w = tid >> 5;
    const int wm = w >> 1, wn = w & 1;
    const int row0 = blockIdx.y * BM, col0 = blockIdx.x * BN;
    const int r = lane & 15, hi = lane >> 4;

    const u32 aBase0 = (u32)(size_t)&AsBuf[0][0], aBase1 = (u32)(size_t)&AsBuf[1][0];
    const u32 bBase0 = (u32)(size_t)&BsBuf[0][0], bBase1 = (u32)(size_t)&BsBuf[1][0];

    v8f acc[2][2];
    acc[0][0] = vzero(); acc[0][1] = vzero(); acc[1][0] = vzero(); acc[1][1] = vzero();

    const int nt = K / BK;
    // Prologue: stream tile 0 into buffer 0
    issue_tile_A(A, row0, K, 0, aBase0, tid);
    issue_tile_B(W, col0, K, 0, bBase0, tid);

    for (int it = 0; it < nt; ++it) {
        const int p = it & 1;
        if (it + 1 < nt) {
            // stream next tile into the alternate buffer, behind the compute
            issue_tile_A(A, row0, K, (it+1)*BK, p ? aBase0 : aBase1, tid);
            issue_tile_B(W, col0, K, (it+1)*BK, p ? bBase0 : bBase1, tid);
            WAIT_ASYNCCNT(6);    // in-order: current tile's 6 copies are done
        } else {
            WAIT_ASYNCCNT(0);
        }
        __syncthreads();

        const u32* Ap = (const u32*)&AsBuf[p][0];
        const u32* Bp = (const u32*)&BsBuf[p][0];
        #pragma unroll
        for (int ks = 0; ks < 2; ++ks) {
            const int k0 = ks * 32;
            Frag a[2], bfr[2];
            #pragma unroll
            for (int mi = 0; mi < 2; ++mi) {
                int ra = wm*32 + mi*16 + r;
                #pragma unroll
                for (int v = 0; v < 8; ++v) {
                    int k = k0 + hi*16 + v*2;
                    a[mi].u[v] = Ap[ra*PITCH32 + (k >> 1)];
                }
            }
            #pragma unroll
            for (int ni = 0; ni < 2; ++ni) {
                int rb = wn*32 + ni*16 + r;
                int kb = (k0 + hi*16) >> 1;
                #pragma unroll
                for (int v = 0; v < 8; ++v)
                    bfr[ni].u[v] = Bp[rb*PITCH32 + kb + v];
            }
            #pragma unroll
            for (int mi = 0; mi < 2; ++mi)
                #pragma unroll
                for (int ni = 0; ni < 2; ++ni)
                    acc[mi][ni] = __builtin_amdgcn_wmma_f32_16x16x32_bf16(
                        false, a[mi].v, false, bfr[ni].v, (short)0, acc[mi][ni], false, false);
        }
        __syncthreads();   // protect buffer p from next iteration's async refill
    }

    const float s = scale_ptr ? *scale_ptr : 1.0f;
    #pragma unroll
    for (int mi = 0; mi < 2; ++mi)
        #pragma unroll
        for (int ni = 0; ni < 2; ++ni) {
            int colT = col0 + wn*32 + ni*16 + r;
            float bv = bias ? bias[colT] : 0.f;
            #pragma unroll
            for (int v = 0; v < 8; ++v) {
                int rowT = row0 + wm*32 + mi*16 + hi*8 + v;
                size_t o = (size_t)rowT * N + colT;
                float val = acc[mi][ni][v] * s + bv + (resid ? resid[o] : 0.f);
                if (Cf) Cf[o] = val;
                if (Cb) Cb[o] = f2bf(val);
            }
        }
}

// ---------------------------------------------------------------------------
// Fused FFN gate+up GEMM: one A tile, two weight tiles in the same K-loop,
// epilogue act = silu(gate) * up -> bf16. Same async double-buffer pipeline
// (8 async copies per tile per wave).
// ---------------------------------------------------------------------------
__global__ __launch_bounds__(256) void ffn_gateup_kernel(
    const u16* __restrict__ A, const u16* __restrict__ Wg, const u16* __restrict__ Wu,
    u16* __restrict__ act, int M, int N, int K)
{
    __shared__ u16 AsBuf[2][BM * PITCH16];
    __shared__ u16 GsBuf[2][BN * PITCH16];
    __shared__ u16 UsBuf[2][BN * PITCH16];

    const int tid = threadIdx.x, lane = tid & 31, w = tid >> 5;
    const int wm = w >> 1, wn = w & 1;
    const int row0 = blockIdx.y * BM, col0 = blockIdx.x * BN;
    const int r = lane & 15, hi = lane >> 4;

    const u32 aB[2] = {(u32)(size_t)&AsBuf[0][0], (u32)(size_t)&AsBuf[1][0]};
    const u32 gB[2] = {(u32)(size_t)&GsBuf[0][0], (u32)(size_t)&GsBuf[1][0]};
    const u32 uB[2] = {(u32)(size_t)&UsBuf[0][0], (u32)(size_t)&UsBuf[1][0]};

    v8f ag[2][2], au[2][2];
    #pragma unroll
    for (int i = 0; i < 2; ++i)
        #pragma unroll
        for (int j = 0; j < 2; ++j) { ag[i][j] = vzero(); au[i][j] = vzero(); }

    const int nt = K / BK;
    issue_tile_A(A,  row0, K, 0, aB[0], tid);
    issue_tile_B(Wg, col0, K, 0, gB[0], tid);
    issue_tile_B(Wu, col0, K, 0, uB[0], tid);

    for (int it = 0; it < nt; ++it) {
        const int p = it & 1;
        if (it + 1 < nt) {
            issue_tile_A(A,  row0, K, (it+1)*BK, aB[1-p], tid);
            issue_tile_B(Wg, col0, K, (it+1)*BK, gB[1-p], tid);
            issue_tile_B(Wu, col0, K, (it+1)*BK, uB[1-p], tid);
            WAIT_ASYNCCNT(8);
        } else {
            WAIT_ASYNCCNT(0);
        }
        __syncthreads();

        const u32* Ap = (const u32*)&AsBuf[p][0];
        const u32* Gp = (const u32*)&GsBuf[p][0];
        const u32* Up = (const u32*)&UsBuf[p][0];
        #pragma unroll
        for (int ks = 0; ks < 2; ++ks) {
            const int k0 = ks * 32;
            Frag a[2], bg[2], bu[2];
            #pragma unroll
            for (int mi = 0; mi < 2; ++mi) {
                int ra = wm*32 + mi*16 + r;
                #pragma unroll
                for (int v = 0; v < 8; ++v) {
                    int k = k0 + hi*16 + v*2;
                    a[mi].u[v] = Ap[ra*PITCH32 + (k >> 1)];
                }
            }
            #pragma unroll
            for (int ni = 0; ni < 2; ++ni) {
                int rb = wn*32 + ni*16 + r;
                int kb = (k0 + hi*16) >> 1;
                #pragma unroll
                for (int v = 0; v < 8; ++v) {
                    bg[ni].u[v] = Gp[rb*PITCH32 + kb + v];
                    bu[ni].u[v] = Up[rb*PITCH32 + kb + v];
                }
            }
            #pragma unroll
            for (int mi = 0; mi < 2; ++mi)
                #pragma unroll
                for (int ni = 0; ni < 2; ++ni) {
                    ag[mi][ni] = __builtin_amdgcn_wmma_f32_16x16x32_bf16(
                        false, a[mi].v, false, bg[ni].v, (short)0, ag[mi][ni], false, false);
                    au[mi][ni] = __builtin_amdgcn_wmma_f32_16x16x32_bf16(
                        false, a[mi].v, false, bu[ni].v, (short)0, au[mi][ni], false, false);
                }
        }
        __syncthreads();
    }

    #pragma unroll
    for (int mi = 0; mi < 2; ++mi)
        #pragma unroll
        for (int ni = 0; ni < 2; ++ni) {
            int colT = col0 + wn*32 + ni*16 + r;
            #pragma unroll
            for (int v = 0; v < 8; ++v) {
                int rowT = row0 + wm*32 + mi*16 + hi*8 + v;
                float g = ag[mi][ni][v];
                float u = au[mi][ni][v];
                act[(size_t)rowT * N + colT] = f2bf(g * sigm(g) * u);
            }
        }
}

// ---------------------------------------------------------------------------
// Host-side orchestration
// ---------------------------------------------------------------------------
extern "C" void kernel_launch(void* const* d_in, const int* in_sizes, int n_in,
                              void* d_out, int out_size, void* d_ws, size_t ws_size,
                              hipStream_t stream) {
    (void)in_sizes; (void)n_in; (void)out_size; (void)ws_size;
    const float* x      = (const float*)d_in[0];
    const float* convw  = (const float*)d_in[1];
    const float* Wd     = (const float*)d_in[2];
    const float* bd     = (const float*)d_in[3];
    const float* Wg     = (const float*)d_in[4];
    const float* bg     = (const float*)d_in[5];
    const float* sscale = (const float*)d_in[6];
    // d_in[7] = Wq : unused by the reference forward
    const float* Wk     = (const float*)d_in[8];
    const float* Wv     = (const float*)d_in[9];
    const float* Wbeta  = (const float*)d_in[10];
    const float* M0     = (const float*)d_in[11];
    const float* Wmem   = (const float*)d_in[12];
    const float* mscale = (const float*)d_in[13];
    const float* Wc     = (const float*)d_in[14];
    const float* bc     = (const float*)d_in[15];
    const float* Wup    = (const float*)d_in[16];
    const float* Wgf    = (const float*)d_in[17];
    const float* Wdn    = (const float*)d_in[18];
    const float* n1     = (const float*)d_in[19];
    const float* n2     = (const float*)d_in[20];
    float* out = (float*)d_out;

    char* ws = (char*)d_ws;
    constexpr size_t MB = 1ull << 20;
    float* h      = (float*)(ws + 0*MB);     // 32MB
    u16*   hb     = (u16*)  (ws + 32*MB);    // 16MB
    float* local  = (float*)(ws + 48*MB);    // 32MB
    float* dpre   = (float*)(ws + 80*MB);    // 32MB  (reused later as memf)
    float* gpre   = (float*)(ws + 112*MB);   // 32MB  (reused later as mixed)
    float* state  = (float*)(ws + 144*MB);   // 32MB
    float* kf     = (float*)(ws + 176*MB);   // 8MB
    float* vf     = (float*)(ws + 184*MB);   // 8MB
    float* betaf  = (float*)(ws + 192*MB);   // 32KB (1MB reserved)
    u16*   memb   = (u16*)  (ws + 193*MB);   // 4MB
    u16*   cat    = (u16*)  (ws + 197*MB);   // 48MB
    u16*   h2b    = (u16*)  (ws + 245*MB);   // 16MB
    u16*   actb   = (u16*)  (ws + 261*MB);   // 64MB
    u16*   WdB    = (u16*)  (ws + 325*MB);
    u16*   WgB    = (u16*)  (ws + 327*MB);
    u16*   WkB    = (u16*)  (ws + 329*MB);
    u16*   WvB    = (u16*)  (ws + 329*MB + 512*1024);
    u16*   WmemB  = (u16*)  (ws + 330*MB);
    u16*   WcB    = (u16*)  (ws + 331*MB);
    u16*   WupB   = (u16*)  (ws + 337*MB);
    u16*   WgfB   = (u16*)  (ws + 345*MB);
    u16*   WdnB   = (u16*)  (ws + 353*MB);
    float* memf   = dpre;   // reuse after state scan
    float* mixed  = gpre;   // reuse after state scan

    auto cvt = [&](const float* src, u16* dst, int n) {
        cvt_bf16_kernel<<<(n + 255)/256, 256, 0, stream>>>(src, dst, n);
    };
    auto gemm = [&](const u16* A, const u16* W, float* Cf, u16* Cb,
                    const float* bias, const float* resid, const float* sc,
                    int M, int N, int K) {
        dim3 grid(N / BN, M / BM);
        gemm_bf16_kernel<<<grid, 256, 0, stream>>>(A, W, Cf, Cb, bias, resid, sc, M, N, K);
    };

    const int BT = B_ * T_;

    // Weight conversion to bf16 (matrix-unit operand precision)
    cvt(Wd,   WdB,   D_*D_);
    cvt(Wg,   WgB,   D_*D_);
    cvt(Wk,   WkB,   DM_*D_);
    cvt(Wv,   WvB,   DM_*D_);
    cvt(Wmem, WmemB, D_*DM_);
    cvt(Wc,   WcB,   D_*3*D_);
    cvt(Wup,  WupB,  DF_*D_);
    cvt(Wgf,  WgfB,  DF_*D_);
    cvt(Wdn,  WdnB,  D_*DF_);

    // h = rmsnorm(x, n1) (f32 + bf16 copy)
    rmsnorm_kernel<<<BT, 256, 0, stream>>>(x, n1, h, hb);

    // local = causal depthwise conv(h)
    dwconv_kernel<<<(BT*D_)/256, 256, 0, stream>>>(h, convw, local);

    // dpre/gpre = h @ Wd^T + bd / h @ Wg^T + bg
    gemm(hb, WdB, dpre, nullptr, bd, nullptr, nullptr, BT, D_, D_);
    gemm(hb, WgB, gpre, nullptr, bg, nullptr, nullptr, BT, D_, D_);

    // state scan (4096 independent chains)
    state_scan_kernel<<<(B_*D_)/256, 256, 0, stream>>>(dpre, gpre, h, sscale, state);

    // k, v, beta
    gemm(hb, WkB, kf, nullptr, nullptr, nullptr, nullptr, BT, DM_, D_);
    gemm(hb, WvB, vf, nullptr, nullptr, nullptr, nullptr, BT, DM_, D_);
    beta_kernel<<<BT/8, 256, 0, stream>>>(h, Wbeta, betaf);

    // DeltaNet memory scan: M resident in 263KB LDS, one WGP
    const int memscan_bytes = MEMSCAN_LDS_FLOATS * (int)sizeof(float);
    hipFuncSetAttribute((const void*)memscan_kernel,
                        hipFuncAttributeMaxDynamicSharedMemorySize, memscan_bytes);
    memscan_kernel<<<1, 1024, memscan_bytes, stream>>>(M0, kf, vf, betaf, memb);

    // mem = (mem_out @ Wmem^T) * mem_scale   (reuses dpre buffer)
    gemm(memb, WmemB, memf, nullptr, nullptr, nullptr, mscale, BT, D_, DM_);

    // cat = bf16([local | state | mem])
    concat_kernel<<<(BT*3*D_)/256, 256, 0, stream>>>(local, state, memf, cat);

    // mixed = x + cat @ Wc^T + bc   (reuses gpre buffer)
    gemm(cat, WcB, mixed, nullptr, bc, x, nullptr, BT, D_, 3*D_);

    // h2 = rmsnorm(mixed, n2) -> bf16
    rmsnorm_kernel<<<BT, 256, 0, stream>>>(mixed, n2, nullptr, h2b);

    // act = silu(h2 @ Wgate^T) * (h2 @ Wup^T) -> bf16 (fused dual GEMM)
    {
        dim3 grid(DF_ / BN, BT / BM);
        ffn_gateup_kernel<<<grid, 256, 0, stream>>>(h2b, WgfB, WupB, actb, BT, DF_, D_);
    }

    // out = mixed + act @ Wdown^T
    gemm(actb, WdnB, out, nullptr, nullptr, mixed, nullptr, BT, D_, DF_);
}
